// TransformerKinematicsDecoder_13056700580053
// MI455X (gfx1250) — matrix operand order repaired and checked by reading, not compiled, not run on hardware
//
#include <hip/hip_runtime.h>

// ---------------------------------------------------------------------------
// Types for CDNA5 WMMA (wave32, 16x16x32 f16 -> f32)
// ---------------------------------------------------------------------------
typedef __attribute__((ext_vector_type(16))) _Float16 v16h;
typedef __attribute__((ext_vector_type(8)))  float    v8f;

#define D_MODEL 256
#define FF_DIM  1024
#define SEQ_L   1536
#define BATCH   4
#define HEADS   8
#define HEADD   32
#define MTOK    (BATCH * SEQ_L)   // 6144

__device__ __forceinline__ float gelu_f(float x) {
    return 0.5f * x * (1.0f + erff(x * 0.70710678118654752440f));
}

// A-fragment (16x32 f16) half-slot -> K index, per ISA table:
// lanes 0-15: slots 0..7 -> K=0..7, slots 8..15 -> K=16..23
// lanes16-31: slots 0..7 -> K=8..15, slots 8..15 -> K=24..31
__device__ __forceinline__ int mapA_k(int slot, int lane) {
    return ((slot < 8) ? 0 : 16) + ((lane >> 4) << 3) + (slot & 7);
}

__device__ __forceinline__ void wait_ds0() {
#if __has_builtin(__builtin_amdgcn_s_wait_dscnt)
    __builtin_amdgcn_s_wait_dscnt(0);
    asm volatile("" ::: "memory");
#else
    asm volatile("s_wait_dscnt 0x0" ::: "memory");
#endif
}

// ---------------------------------------------------------------------------
// fp32 -> f16 weight conversion
// ---------------------------------------------------------------------------
__global__ void f2h_kernel(const float* __restrict__ in, _Float16* __restrict__ out, int n) {
    int i = blockIdx.x * blockDim.x + threadIdx.x;
    if (i < n) out[i] = (_Float16)in[i];
}

// ---------------------------------------------------------------------------
// Per-batch active-channel order + counts
// ---------------------------------------------------------------------------
__global__ void build_meta(const int* __restrict__ mask, int* __restrict__ ord,
                           int* __restrict__ cnt, int* __restrict__ vlen) {
    int b = threadIdx.x;
    if (b >= BATCH) return;
    int c = 0;
    for (int i = 0; i < 96; ++i)
        if (mask[b * 96 + i] != 0) ord[b * 96 + c++] = i;
    if (c < 1) c = 1;            // reference guarantees >=1 active
    cnt[b]  = c;
    vlen[b] = 16 * c;
}

// ---------------------------------------------------------------------------
// Embedding: x[b,p,d] = v*val_w + val_b + ch_emb + t_emb + sess
// ---------------------------------------------------------------------------
__global__ void embed_kernel(const float* __restrict__ sbp, const int* __restrict__ cnt,
                             const int* __restrict__ ord, const float* __restrict__ days,
                             const float* __restrict__ ch_emb, const float* __restrict__ t_emb,
                             const float* __restrict__ val_w, const float* __restrict__ val_b,
                             const float* __restrict__ sess_w, const float* __restrict__ sess_b,
                             float* __restrict__ x) {
    int tok = blockIdx.x;                 // 0..6143
    int b = tok / SEQ_L, p = tok % SEQ_L;
    int d = threadIdx.x;
    int count = cnt[b];
    bool valid = p < 16 * count;
    int t_id = valid ? (p / count) : 0;
    int ch   = valid ? ord[b * 96 + (p % count)] : 0;
    float value = valid ? sbp[b * SEQ_L + t_id * 96 + ch] : 0.0f;
    float sd = days[b];
    x[(size_t)tok * D_MODEL + d] =
        value * val_w[d] + val_b[d] + ch_emb[ch * D_MODEL + d] + t_emb[t_id * D_MODEL + d]
        + sd * sess_w[d] + sess_b[d];
}

// ---------------------------------------------------------------------------
// Wave-per-row LayerNorm over D=256. 8 waves/block, 8 floats/lane.
// HALF=1 -> f16 output (GEMM A operand); HALF=0 -> fp32 output.
// ---------------------------------------------------------------------------
template <int HALF>
__global__ void ln_rows(const float* __restrict__ X, const float* __restrict__ g,
                        const float* __restrict__ bta, _Float16* __restrict__ o16,
                        float* __restrict__ o32, int M) {
    int lane = threadIdx.x & 31;
    int row  = blockIdx.x * 8 + (threadIdx.x >> 5);
    if (row >= M) return;
    const float* xr = X + (size_t)row * D_MODEL;
    const float4* xv = reinterpret_cast<const float4*>(xr);
    float4 v0 = xv[lane * 2 + 0];
    float4 v1 = xv[lane * 2 + 1];
    float vv[8] = {v0.x, v0.y, v0.z, v0.w, v1.x, v1.y, v1.z, v1.w};
    float s = 0.f, q = 0.f;
#pragma unroll
    for (int j = 0; j < 8; ++j) { s += vv[j]; q += vv[j] * vv[j]; }
#pragma unroll
    for (int off = 1; off < 32; off <<= 1) {
        s += __shfl_xor(s, off, 32);
        q += __shfl_xor(q, off, 32);
    }
    float mu  = s * (1.0f / 256.0f);
    float var = q * (1.0f / 256.0f) - mu * mu;
    float rs  = rsqrtf(var + 1e-5f);
    int d0 = lane * 8;
#pragma unroll
    for (int j = 0; j < 8; ++j) {
        float y = (vv[j] - mu) * rs * g[d0 + j] + bta[d0 + j];
        if (HALF) o16[(size_t)row * D_MODEL + d0 + j] = (_Float16)y;
        else      o32[(size_t)row * D_MODEL + d0 + j] = y;
    }
}

// ---------------------------------------------------------------------------
// WMMA GEMM: C[M,N] = A16[M,K] @ B16[K,N] + bias, optional GELU,
// output to f16 (HALF_OUT) or fp32 residual-add (RESID) or plain fp32.
// Block: 128 threads (4 waves). Block tile 64x64, K-step 32.
// ---------------------------------------------------------------------------
template <int ACT, int HALF_OUT, int RESID>
__global__ void gemm16(const _Float16* __restrict__ A, const _Float16* __restrict__ B,
                       const float* __restrict__ bias, float* __restrict__ Cf,
                       _Float16* __restrict__ Ch, int M, int N, int K) {
    __shared__ __attribute__((aligned(16))) _Float16 As[64 * 32];
    __shared__ __attribute__((aligned(16))) _Float16 Bs[32 * 64];
    const int tid  = threadIdx.x;
    const int lane = tid & 31;
    const int wave = tid >> 5;
    const int mb = blockIdx.y * 64;
    const int nb = blockIdx.x * 64;
    v8f acc[4] = {};

    for (int kb = 0; kb < K; kb += 32) {
        // Stage A (64x32 halves) and B (32x64 halves) via 16B vector ops.
#pragma unroll
        for (int i = 0; i < 2; ++i) {
            int s = tid + i * 128;                 // 0..255
            int ar = s >> 2, ao = (s & 3) * 8;
            *reinterpret_cast<uint4*>(&As[ar * 32 + ao]) =
                *reinterpret_cast<const uint4*>(&A[(size_t)(mb + ar) * K + kb + ao]);
            int br = s >> 3, bo = (s & 7) * 8;
            *reinterpret_cast<uint4*>(&Bs[br * 64 + bo]) =
                *reinterpret_cast<const uint4*>(&B[(size_t)(kb + br) * N + nb + bo]);
        }
        if (kb + 32 < K) {
            // hint next K-tile into cache (global_prefetch_b8)
            __builtin_prefetch(&A[(size_t)(mb + (tid >> 1)) * K + kb + 32], 0, 1);
            __builtin_prefetch(&B[(size_t)(kb + 32 + (tid & 31)) * N + nb], 0, 1);
        }
        __syncthreads();

        v16h af;
        int arow = wave * 16 + (lane & 15);
#pragma unroll
        for (int h = 0; h < 16; ++h) af[h] = As[arow * 32 + ((lane >> 4) << 4) + h];
#pragma unroll
        for (int nt = 0; nt < 4; ++nt) {
            v16h bf;
            int col = nt * 16 + (lane & 15);
#pragma unroll
            for (int h = 0; h < 16; ++h)
                bf[h] = Bs[(((lane >> 4) << 4) + h) * 64 + col];
            acc[nt] = __builtin_amdgcn_wmma_f32_16x16x32_f16(
                false, af, false, bf, (short)0, acc[nt], false, false);
        }
        __syncthreads();
    }

#pragma unroll
    for (int nt = 0; nt < 4; ++nt) {
#pragma unroll
        for (int i = 0; i < 8; ++i) {
            int m   = i + ((lane >> 4) << 3);
            int row = mb + wave * 16 + m;
            int col = nb + nt * 16 + (lane & 15);
            float v = acc[nt][i] + bias[col];
            if (ACT == 1) v = gelu_f(v);
            if (RESID)          Cf[(size_t)row * N + col] += v;
            else if (HALF_OUT)  Ch[(size_t)row * N + col] = (_Float16)v;
            else                Cf[(size_t)row * N + col] = v;
        }
    }
}

// ---------------------------------------------------------------------------
// Flash attention, wave32 WMMA. One wave = 16 query rows of one (b,h).
// HD=32 == one 16x16x32 k-slice. Keys processed 32 at a time (two S tiles).
// Grid: 384 blocks (96 per batch item) x 256 threads (8 waves).
// ---------------------------------------------------------------------------
__global__ void attn_kernel(const _Float16* __restrict__ Q, const _Float16* __restrict__ Kk,
                            const _Float16* __restrict__ V, _Float16* __restrict__ O,
                            const int* __restrict__ vlen) {
    __shared__ _Float16 pbuf[8][16 * 32];   // per-wave private P tile
    const int tid  = threadIdx.x;
    const int lane = tid & 31;
    const int wave = tid >> 5;
    const int grp  = lane >> 4;             // half-wave group
    const int col  = lane & 15;
    const int b    = blockIdx.x / 96;
    const int wix  = (blockIdx.x % 96) * 8 + wave;   // 0..767 = h*96 + qb
    const int h    = wix / 96;
    const int qb   = wix % 96;
    const int VL   = vlen[b];
    const float scale = 0.17677669529663688110f;     // 1/sqrt(32)

    // Q fragment (A layout), rows qb*16..+15
    v16h qf;
    {
        int row = qb * 16 + col;
        size_t base = (((size_t)b * SEQ_L + row) * HEADS + h) * HEADD;
#pragma unroll
        for (int t = 0; t < 16; ++t) qf[t] = Q[base + (grp << 4) + t];
    }

    v8f a0 = {}, a1 = {};
    float mi[8], li[8];
#pragma unroll
    for (int i = 0; i < 8; ++i) { mi[i] = -1e30f; li[i] = 0.0f; }

    for (int kb = 0; kb < VL; kb += 32) {
        // K^T fragments: column=key, Kdim=hd -> contiguous 16 halves per lane
        int key_a = kb + col;
        int key_b = key_a + 16;
        int kr0 = min(key_a, SEQ_L - 1);
        int kr1 = min(key_b, SEQ_L - 1);
        v16h kf0 = *reinterpret_cast<const v16h*>(
            &Kk[(((size_t)b * SEQ_L + kr0) * HEADS + h) * HEADD + grp * 16]);
        v16h kf1 = *reinterpret_cast<const v16h*>(
            &Kk[(((size_t)b * SEQ_L + kr1) * HEADS + h) * HEADD + grp * 16]);

        v8f z = {};
        v8f s0 = __builtin_amdgcn_wmma_f32_16x16x32_f16(false, qf, false, kf0, (short)0, z, false, false);
        v8f s1 = __builtin_amdgcn_wmma_f32_16x16x32_f16(false, qf, false, kf1, (short)0, z, false, false);

        bool ok0 = key_a < VL, ok1 = key_b < VL;
        float mt[8];
#pragma unroll
        for (int i = 0; i < 8; ++i) {
            float x0 = ok0 ? s0[i] * scale : -1e30f;
            float x1 = ok1 ? s1[i] * scale : -1e30f;
            s0[i] = x0; s1[i] = x1;
            mt[i] = fmaxf(x0, x1);
        }
#pragma unroll
        for (int off = 1; off < 16; off <<= 1)
#pragma unroll
            for (int i = 0; i < 8; ++i) mt[i] = fmaxf(mt[i], __shfl_xor(mt[i], off, 32));

        float al[8], ls[8];
#pragma unroll
        for (int i = 0; i < 8; ++i) {
            float mn = fmaxf(mi[i], mt[i]);
            al[i] = __expf(mi[i] - mn);
            float p0 = __expf(s0[i] - mn);
            float p1 = __expf(s1[i] - mn);
            s0[i] = p0; s1[i] = p1;
            ls[i] = p0 + p1;
            mi[i] = mn;
        }
#pragma unroll
        for (int off = 1; off < 16; off <<= 1)
#pragma unroll
            for (int i = 0; i < 8; ++i) ls[i] += __shfl_xor(ls[i], off, 32);
#pragma unroll
        for (int i = 0; i < 8; ++i) {
            li[i] = li[i] * al[i] + ls[i];
            a0[i] *= al[i];
            a1[i] *= al[i];
        }

        // C-layout -> A-layout via wave-private LDS
        _Float16* pw = &pbuf[wave][0];
#pragma unroll
        for (int i = 0; i < 8; ++i) {
            int m = i + (grp << 3);
            pw[m * 32 + col]      = (_Float16)s0[i];
            pw[m * 32 + 16 + col] = (_Float16)s1[i];
        }
        wait_ds0();
        v16h pf;
        {
            int row = col;
#pragma unroll
            for (int t = 0; t < 16; ++t) pf[t] = pw[row * 32 + (grp << 4) + t];
        }

        // V fragments (B layout): column = hd, Kdim = key-in-chunk
        v16h vf0, vf1;
#pragma unroll
        for (int t = 0; t < 16; ++t) {
            int krow = min(kb + (grp << 4) + t, SEQ_L - 1);
            size_t bv = (((size_t)b * SEQ_L + krow) * HEADS + h) * HEADD;
            vf0[t] = V[bv + col];
            vf1[t] = V[bv + 16 + col];
        }
        a0 = __builtin_amdgcn_wmma_f32_16x16x32_f16(false, pf, false, vf0, (short)0, a0, false, false);
        a1 = __builtin_amdgcn_wmma_f32_16x16x32_f16(false, pf, false, vf1, (short)0, a1, false, false);
    }

#pragma unroll
    for (int i = 0; i < 8; ++i) {
        int m   = i + (grp << 3);
        int row = qb * 16 + m;
        float inv = 1.0f / li[i];
        size_t bo = (((size_t)b * SEQ_L + row) * HEADS + h) * HEADD;
        O[bo + col]      = (_Float16)(a0[i] * inv);
        O[bo + 16 + col] = (_Float16)(a1[i] * inv);
    }
}

// ---------------------------------------------------------------------------
// Masked mean pool over valid tokens
// ---------------------------------------------------------------------------
__global__ void pool_kernel(const float* __restrict__ xln, const int* __restrict__ vlen,
                            float* __restrict__ pooled) {
    int b = blockIdx.x, d = threadIdx.x;
    int VL = vlen[b];
    const float* p = xln + (size_t)b * SEQ_L * D_MODEL + d;
    float acc = 0.0f;
    for (int l = 0; l < VL; ++l) acc += p[(size_t)l * D_MODEL];
    pooled[b * D_MODEL + d] = acc / (float)VL;
}

// ---------------------------------------------------------------------------
// Head: GEMV 256->1024 + GELU, then 1024->2
// ---------------------------------------------------------------------------
__global__ void head_ff1(const float* __restrict__ hh, const float* __restrict__ hw1,
                         const float* __restrict__ hb1, float* __restrict__ g1) {
    int idx = blockIdx.x * blockDim.x + threadIdx.x;   // 4096
    int b = idx >> 10, f = idx & 1023;
    float acc = hb1[f];
    for (int k = 0; k < D_MODEL; ++k) acc += hh[b * D_MODEL + k] * hw1[k * FF_DIM + f];
    g1[idx] = gelu_f(acc);
}

__global__ void head_out(const float* __restrict__ g1, const float* __restrict__ hw2,
                         const float* __restrict__ hb2, float* __restrict__ out) {
    int t = threadIdx.x;
    if (t >= BATCH * 2) return;
    int b = t >> 1, o = t & 1;
    float acc = hb2[o];
    for (int k = 0; k < FF_DIM; ++k) acc += g1[b * FF_DIM + k] * hw2[k * 2 + o];
    out[b * 2 + o] = acc;
}

// ---------------------------------------------------------------------------
// Host launcher
// ---------------------------------------------------------------------------
extern "C" void kernel_launch(void* const* d_in, const int* in_sizes, int n_in,
                              void* d_out, int out_size, void* d_ws, size_t ws_size,
                              hipStream_t stream) {
    (void)in_sizes; (void)n_in; (void)out_size; (void)ws_size;
    const float* sbp    = (const float*)d_in[0];
    const int*   mask   = (const int*)  d_in[1];
    const float* days   = (const float*)d_in[2];
    const float* ch_emb = (const float*)d_in[3];
    const float* t_emb  = (const float*)d_in[4];
    const float* val_w  = (const float*)d_in[5];
    const float* val_b  = (const float*)d_in[6];
    const float* sess_w = (const float*)d_in[7];
    const float* sess_b = (const float*)d_in[8];
    const float* ln1_g  = (const float*)d_in[9];
    const float* ln1_b  = (const float*)d_in[10];
    const float* wq     = (const float*)d_in[11];
    const float* bq     = (const float*)d_in[12];
    const float* wk     = (const float*)d_in[13];
    const float* bk     = (const float*)d_in[14];
    const float* wv     = (const float*)d_in[15];
    const float* bv     = (const float*)d_in[16];
    const float* wo     = (const float*)d_in[17];
    const float* bo     = (const float*)d_in[18];
    const float* ln2_g  = (const float*)d_in[19];
    const float* ln2_b  = (const float*)d_in[20];
    const float* w1     = (const float*)d_in[21];
    const float* b1     = (const float*)d_in[22];
    const float* w2     = (const float*)d_in[23];
    const float* b2     = (const float*)d_in[24];
    const float* enc_g  = (const float*)d_in[25];
    const float* enc_b  = (const float*)d_in[26];
    const float* hl_g   = (const float*)d_in[27];
    const float* hl_b   = (const float*)d_in[28];
    const float* hw1    = (const float*)d_in[29];
    const float* hb1    = (const float*)d_in[30];
    const float* hw2    = (const float*)d_in[31];
    const float* hb2    = (const float*)d_in[32];
    float* out = (float*)d_out;

    char* wsp = (char*)d_ws;
    auto take = [&wsp](size_t bytes) -> void* {
        void* p = (void*)wsp;
        wsp += (bytes + 255) & ~(size_t)255;
        return p;
    };

    const int M = MTOK;
    float*     x     = (float*)    take((size_t)M * D_MODEL * 4);
    _Float16*  h16   = (_Float16*) take((size_t)M * D_MODEL * 2);
    _Float16*  q16   = (_Float16*) take((size_t)M * D_MODEL * 2);
    _Float16*  k16   = (_Float16*) take((size_t)M * D_MODEL * 2);
    _Float16*  v16   = (_Float16*) take((size_t)M * D_MODEL * 2);
    _Float16*  o16   = (_Float16*) take((size_t)M * D_MODEL * 2);
    _Float16*  ff16  = (_Float16*) take((size_t)M * FF_DIM * 2);
    float*     xln   = (float*)    take((size_t)M * D_MODEL * 4);
    _Float16*  wq16  = (_Float16*) take((size_t)3 * D_MODEL * D_MODEL * 2);
    _Float16*  wk16  = (_Float16*) take((size_t)3 * D_MODEL * D_MODEL * 2);
    _Float16*  wv16  = (_Float16*) take((size_t)3 * D_MODEL * D_MODEL * 2);
    _Float16*  wo16  = (_Float16*) take((size_t)3 * D_MODEL * D_MODEL * 2);
    _Float16*  w116  = (_Float16*) take((size_t)3 * D_MODEL * FF_DIM * 2);
    _Float16*  w216  = (_Float16*) take((size_t)3 * FF_DIM * D_MODEL * 2);
    float*     pooled= (float*)    take((size_t)BATCH * D_MODEL * 4);
    float*     hh    = (float*)    take((size_t)BATCH * D_MODEL * 4);
    float*     g1    = (float*)    take((size_t)BATCH * FF_DIM * 4);
    int*       cnt   = (int*)      take(BATCH * 4);
    int*       vlen  = (int*)      take(BATCH * 4);
    int*       ordb  = (int*)      take(BATCH * 96 * 4);

    // Weight conversion fp32 -> f16
    {
        int n = 3 * D_MODEL * D_MODEL;
        f2h_kernel<<<(n + 255) / 256, 256, 0, stream>>>(wq, wq16, n);
        f2h_kernel<<<(n + 255) / 256, 256, 0, stream>>>(wk, wk16, n);
        f2h_kernel<<<(n + 255) / 256, 256, 0, stream>>>(wv, wv16, n);
        f2h_kernel<<<(n + 255) / 256, 256, 0, stream>>>(wo, wo16, n);
        int nf = 3 * D_MODEL * FF_DIM;
        f2h_kernel<<<(nf + 255) / 256, 256, 0, stream>>>(w1, w116, nf);
        f2h_kernel<<<(nf + 255) / 256, 256, 0, stream>>>(w2, w216, nf);
    }

    build_meta<<<1, 32, 0, stream>>>(mask, ordb, cnt, vlen);
    embed_kernel<<<MTOK, 256, 0, stream>>>(sbp, cnt, ordb, days, ch_emb, t_emb,
                                           val_w, val_b, sess_w, sess_b, x);

    dim3 gQ(D_MODEL / 64, M / 64);      // (4, 96)
    dim3 gF(FF_DIM / 64, M / 64);       // (16, 96)

    for (int l = 0; l < 3; ++l) {
        const size_t wDD = (size_t)l * D_MODEL * D_MODEL;
        const size_t wDF = (size_t)l * D_MODEL * FF_DIM;
        ln_rows<1><<<M / 8, 256, 0, stream>>>(x, ln1_g + l * D_MODEL, ln1_b + l * D_MODEL,
                                              h16, nullptr, M);
        gemm16<0, 1, 0><<<gQ, 128, 0, stream>>>(h16, wq16 + wDD, bq + l * D_MODEL,
                                                nullptr, q16, M, D_MODEL, D_MODEL);
        gemm16<0, 1, 0><<<gQ, 128, 0, stream>>>(h16, wk16 + wDD, bk + l * D_MODEL,
                                                nullptr, k16, M, D_MODEL, D_MODEL);
        gemm16<0, 1, 0><<<gQ, 128, 0, stream>>>(h16, wv16 + wDD, bv + l * D_MODEL,
                                                nullptr, v16, M, D_MODEL, D_MODEL);
        attn_kernel<<<BATCH * 96, 256, 0, stream>>>(q16, k16, v16, o16, vlen);
        gemm16<0, 0, 1><<<gQ, 128, 0, stream>>>(o16, wo16 + wDD, bo + l * D_MODEL,
                                                x, nullptr, M, D_MODEL, D_MODEL);
        ln_rows<1><<<M / 8, 256, 0, stream>>>(x, ln2_g + l * D_MODEL, ln2_b + l * D_MODEL,
                                              h16, nullptr, M);
        gemm16<1, 1, 0><<<gF, 128, 0, stream>>>(h16, w116 + wDF, b1 + l * FF_DIM,
                                                nullptr, ff16, M, FF_DIM, D_MODEL);
        gemm16<0, 0, 1><<<gQ, 128, 0, stream>>>(ff16, w216 + wDF, b2 + l * D_MODEL,
                                                x, nullptr, M, D_MODEL, FF_DIM);
    }

    ln_rows<0><<<M / 8, 256, 0, stream>>>(x, enc_g, enc_b, nullptr, xln, M);
    pool_kernel<<<BATCH, 256, 0, stream>>>(xln, vlen, pooled);
    ln_rows<0><<<1, 256, 0, stream>>>(pooled, hl_g, hl_b, nullptr, hh, BATCH);
    head_ff1<<<(BATCH * FF_DIM) / 256, 256, 0, stream>>>(hh, hw1, hb1, g1);
    head_out<<<1, 64, 0, stream>>>(g1, hw2, hb2, out);
}
